// Encoder_Flows_6150393168184
// MI455X (gfx1250) — matrix operand order repaired and checked by reading, not compiled, not run on hardware
//
#include <hip/hip_runtime.h>
#include <stdint.h>

// ---------------------------------------------------------------------------
// 3-layer dense GCN on MI455X (gfx1250).
//   out_l = diag(dinv) * A^T * diag(dinv) * (X_l W_l) + b_l
// All six GEMMs on v_wmma_f32_16x16x32_bf16 (f32 accumulate), LDS double-
// buffered; the bf16 A-operand tiles stream via GLOBAL_LOAD_ASYNC_TO_LDS_B128
// (ASYNCcnt) so the dominant operand never passes through VGPRs.
// ---------------------------------------------------------------------------

typedef __attribute__((ext_vector_type(16))) __bf16 bf16x16;
typedef __attribute__((ext_vector_type(8)))  float  f32x8;

__device__ __forceinline__ unsigned short f2bf(float f) {
    // round-to-nearest-even fp32 -> bf16 (bit trick)
    unsigned int u = __float_as_uint(f);
    u += 0x7FFFu + ((u >> 16) & 1u);
    return (unsigned short)(u >> 16);
}
__device__ __forceinline__ unsigned int pk2(float lo, float hi) {
    return (unsigned int)f2bf(lo) | ((unsigned int)f2bf(hi) << 16);
}

// dinv[b,c] = rsqrt(sum_r A[b,r,c])  (column sums, coalesced across threads)
__global__ __launch_bounds__(256)
void degree_rsqrt_kernel(const float* __restrict__ A, float* __restrict__ dinv, int n) {
    const int c = blockIdx.x * blockDim.x + threadIdx.x;
    const int b = blockIdx.y;
    const float* Ab = A + (size_t)b * n * n;
    float s = 0.f;
    for (int r = 0; r < n; ++r) s += Ab[(size_t)r * n + c];
    dinv[(size_t)b * n + c] = (s > 0.f) ? __frsqrt_rn(s) : 0.f;
}

// At[b,c,r] = bf16(A[b,r,c])  — LDS-tiled 32x32 transpose, coalesced both sides
__global__ __launch_bounds__(256)
void transpose_to_bf16_kernel(const float* __restrict__ A, unsigned short* __restrict__ At, int n) {
    __shared__ unsigned short T[32][33];
    const int b  = blockIdx.z;
    const int r0 = blockIdx.y * 32, c0 = blockIdx.x * 32;
    const float* Ab = A + (size_t)b * n * n;
    unsigned short* Tb = At + (size_t)b * n * n;
    const int tx = threadIdx.x & 31, ty = threadIdx.x >> 5;   // ty in 0..7
#pragma unroll
    for (int j = 0; j < 32; j += 8)
        T[tx][ty + j] = f2bf(Ab[(size_t)(r0 + ty + j) * n + c0 + tx]);
    __syncthreads();
#pragma unroll
    for (int j = 0; j < 32; j += 8)
        Tb[(size_t)(c0 + ty + j) * n + r0 + tx] = T[ty + j][tx];
}

__global__ __launch_bounds__(256)
void f32_to_bf16_kernel(const float* __restrict__ x, unsigned short* __restrict__ y, int n) {
    const int i = blockIdx.x * 256 + threadIdx.x;
    if (i < n) y[i] = f2bf(x[i]);
}

template <bool OBF16>
__device__ __forceinline__ void store_tile(void* Cb, int ldc, int row0, int col,
                                           f32x8 acc, const float* scl, float bv) {
#pragma unroll
    for (int r = 0; r < 8; ++r) {
        const float val = scl[row0 + r] * acc[r] + bv;
        if (OBF16) ((unsigned short*)Cb)[(size_t)(row0 + r) * ldc + col] = f2bf(val);
        else       ((float*)Cb)[(size_t)(row0 + r) * ldc + col] = val;
    }
}

// C[m,n] = scale[m] * sum_k op(A)[m,k] * B[k,n]  (+ bias[n])
//   AMODE 0: A fp32, row-major M x K         (layer-1 feature GEMM on flows)
//   AMODE 1: A fp32, physically K x M (A^T)  (fallback aggregation path)
//   AMODE 2: A bf16, row-major M x K         (main path; staged with
//            global_load_async_to_lds_b128, tracked by ASYNCcnt)
// B is always bf16 (K x N).  Output fp32 or bf16 via OBF16.
// Block = 256 threads (8 waves); macro-tile 64(M) x 128(N); each wave owns a
// 32x32 register tile = 4 v_wmma_f32_16x16x32_bf16 / K-step.  LDS is double-
// buffered: tile kt+32 is prefetched while tile kt is consumed -> one barrier
// per K-step.  Grid: (N/128, M/64, batch); all dims multiples of tile sizes.
template <int AMODE, bool OBF16>
__global__ __launch_bounds__(256)
void gemm_rowscale_bias(const void* __restrict__ Av, size_t sA, int lda,
                        const unsigned short* __restrict__ Bm, size_t sB, int ldb,
                        const float* __restrict__ scale, size_t sScale,
                        const float* __restrict__ bias,
                        void* __restrict__ Cv, size_t sC, int ldc, int K) {
    // rows padded to 40 ushorts (80 B) so every gather is a 16B-aligned b128
    __shared__ __align__(16) unsigned short As [2][64][40];    // [buf][m][k]
    __shared__ __align__(16) unsigned short BsT[2][128][40];   // [buf][n][k]

    const int bz = blockIdx.z;
    const unsigned short* Bb = Bm + (size_t)bz * sB;
    const float* scl = scale + (size_t)bz * sScale;

    const int m0 = blockIdx.y * 64;
    const int n0 = blockIdx.x * 128;

    const int tid  = threadIdx.x;
    const int lane = tid & 31;
    const int wave = tid >> 5;           // 0..7
    const int half = lane >> 4;          // 0/1
    const int lr   = lane & 15;
    const int wm   = (wave >> 2) * 32;   // wave M offset in macro-tile
    const int wn   = (wave & 3) * 32;    // wave N offset in macro-tile

    const f32x8 z = {0.f,0.f,0.f,0.f,0.f,0.f,0.f,0.f};
    f32x8 acc00 = z, acc01 = z, acc10 = z, acc11 = z;

    // staging coordinates
    const int amr = tid >> 2;            // 0..63   A row        (AMODE 0/2)
    const int akc = (tid & 3) * 8;       // 0..24   A k-chunk    (AMODE 0/2)
    const int tkr = tid >> 3;            // 0..31   global k row (AMODE 1)
    const int tmc = (tid & 7) * 8;       // 0..56   m-chunk      (AMODE 1)
    const int bkr = tid >> 4;            // 0..15   B k row
    const int bnc = (tid & 15) * 8;      // 0..120  B n-chunk

    // ---- synchronous prologue: stage tile kt=0 into buffer 0 ----
    {
        if (AMODE == 2) {
            const unsigned short* Ab = (const unsigned short*)Av + (size_t)bz * sA;
            *(uint4*)&As[0][amr][akc] =
                *(const uint4*)(Ab + (size_t)(m0 + amr) * lda + akc);
        } else if (AMODE == 0) {
            const float* Ab = (const float*)Av + (size_t)bz * sA;
            const float4 v0 = *(const float4*)(Ab + (size_t)(m0 + amr) * lda + akc);
            const float4 v1 = *(const float4*)(Ab + (size_t)(m0 + amr) * lda + akc + 4);
            uint4 p;
            p.x = pk2(v0.x, v0.y); p.y = pk2(v0.z, v0.w);
            p.z = pk2(v1.x, v1.y); p.w = pk2(v1.z, v1.w);
            *(uint4*)&As[0][amr][akc] = p;
        } else {
            const float* Ab = (const float*)Av + (size_t)bz * sA;
            const float4 v0 = *(const float4*)(Ab + (size_t)tkr * lda + m0 + tmc);
            const float4 v1 = *(const float4*)(Ab + (size_t)tkr * lda + m0 + tmc + 4);
            As[0][tmc + 0][tkr] = f2bf(v0.x); As[0][tmc + 1][tkr] = f2bf(v0.y);
            As[0][tmc + 2][tkr] = f2bf(v0.z); As[0][tmc + 3][tkr] = f2bf(v0.w);
            As[0][tmc + 4][tkr] = f2bf(v1.x); As[0][tmc + 5][tkr] = f2bf(v1.y);
            As[0][tmc + 6][tkr] = f2bf(v1.z); As[0][tmc + 7][tkr] = f2bf(v1.w);
        }
        union U8 { uint4 q; unsigned short u[8]; } q0, q1;
        q0.q = *(const uint4*)(Bb + (size_t)bkr * ldb + n0 + bnc);
        q1.q = *(const uint4*)(Bb + (size_t)(bkr + 16) * ldb + n0 + bnc);
#pragma unroll
        for (int j = 0; j < 8; ++j) {
            BsT[0][bnc + j][bkr]      = q0.u[j];
            BsT[0][bnc + j][bkr + 16] = q1.u[j];
        }
    }
    __syncthreads();

    for (int kt = 0; kt < K; kt += 32) {
        const int cur = (kt >> 5) & 1;
        const int nxt = cur ^ 1;
        const bool pre = (kt + 32) < K;          // uniform across the block
        const int kn = kt + 32;

        // -- early: launch next-tile traffic so it overlaps the WMMAs --
        uint4 qb0 = {}, qb1 = {};
        float4 av0 = {}, av1 = {};
        if (pre) {
            qb0 = *(const uint4*)(Bb + (size_t)(kn + bkr) * ldb + n0 + bnc);
            qb1 = *(const uint4*)(Bb + (size_t)(kn + bkr + 16) * ldb + n0 + bnc);
            if (AMODE == 2) {
                // pure aligned copy -> async global->LDS DMA (ASYNCcnt)
                const unsigned short* g = (const unsigned short*)Av + (size_t)bz * sA +
                                          (size_t)(m0 + amr) * lda + kn + akc;
                const unsigned lds = (unsigned)(uintptr_t)&As[nxt][amr][akc];
                asm volatile("global_load_async_to_lds_b128 %0, %1, off"
                             :: "v"(lds), "v"((unsigned long long)(uintptr_t)g)
                             : "memory");
            } else if (AMODE == 0) {
                const float* Ab = (const float*)Av + (size_t)bz * sA;
                av0 = *(const float4*)(Ab + (size_t)(m0 + amr) * lda + kn + akc);
                av1 = *(const float4*)(Ab + (size_t)(m0 + amr) * lda + kn + akc + 4);
            } else {
                const float* Ab = (const float*)Av + (size_t)bz * sA;
                av0 = *(const float4*)(Ab + (size_t)(kn + tkr) * lda + m0 + tmc);
                av1 = *(const float4*)(Ab + (size_t)(kn + tkr) * lda + m0 + tmc + 4);
            }
        }

        // -- gather current tile (documented gfx1250 16-bit operand layouts) --
        // A operand: lane<16 holds K {0..7,16..23} of row M=lr; lane>=16 K {8..15,24..31}
        // B operand: lane holds column N=lr, same K order as A: lane<16 K {0..7,16..23}; lane>=16 K {8..15,24..31}
        union V16 { bf16x16 v; uint4 q[2]; } a0, a1, b0, b1;
        const unsigned short* pa0 = &As [cur][wm      + lr][half * 8];
        const unsigned short* pa1 = &As [cur][wm + 16 + lr][half * 8];
        const unsigned short* pb0 = &BsT[cur][wn      + lr][half * 8];
        const unsigned short* pb1 = &BsT[cur][wn + 16 + lr][half * 8];
        a0.q[0] = *(const uint4*)pa0;  a0.q[1] = *(const uint4*)(pa0 + 16);
        a1.q[0] = *(const uint4*)pa1;  a1.q[1] = *(const uint4*)(pa1 + 16);
        b0.q[0] = *(const uint4*)pb0;  b0.q[1] = *(const uint4*)(pb0 + 16);
        b1.q[0] = *(const uint4*)pb1;  b1.q[1] = *(const uint4*)(pb1 + 16);

        acc00 = __builtin_amdgcn_wmma_f32_16x16x32_bf16(false, a0.v, false, b0.v, (short)0, acc00, false, false);
        acc01 = __builtin_amdgcn_wmma_f32_16x16x32_bf16(false, a0.v, false, b1.v, (short)0, acc01, false, false);
        acc10 = __builtin_amdgcn_wmma_f32_16x16x32_bf16(false, a1.v, false, b0.v, (short)0, acc10, false, false);
        acc11 = __builtin_amdgcn_wmma_f32_16x16x32_bf16(false, a1.v, false, b1.v, (short)0, acc11, false, false);

        // -- late: commit next tile into the other LDS buffer --
        if (pre) {
            union U8 { uint4 q; unsigned short u[8]; } u0, u1;
            u0.q = qb0; u1.q = qb1;
#pragma unroll
            for (int j = 0; j < 8; ++j) {
                BsT[nxt][bnc + j][bkr]      = u0.u[j];
                BsT[nxt][bnc + j][bkr + 16] = u1.u[j];
            }
            if (AMODE == 0) {
                uint4 p;
                p.x = pk2(av0.x, av0.y); p.y = pk2(av0.z, av0.w);
                p.z = pk2(av1.x, av1.y); p.w = pk2(av1.z, av1.w);
                *(uint4*)&As[nxt][amr][akc] = p;
            } else if (AMODE == 1) {
                As[nxt][tmc + 0][tkr] = f2bf(av0.x); As[nxt][tmc + 1][tkr] = f2bf(av0.y);
                As[nxt][tmc + 2][tkr] = f2bf(av0.z); As[nxt][tmc + 3][tkr] = f2bf(av0.w);
                As[nxt][tmc + 4][tkr] = f2bf(av1.x); As[nxt][tmc + 5][tkr] = f2bf(av1.y);
                As[nxt][tmc + 6][tkr] = f2bf(av1.z); As[nxt][tmc + 7][tkr] = f2bf(av1.w);
            }
        }
        // async copies must have landed in LDS before any wave passes the barrier
        asm volatile("s_wait_asynccnt 0x0" ::: "memory");
        __syncthreads();
    }

    // ---- epilogue: C layout VGPR i -> M = i + 8*half, N = lr ----
    void* Cb = OBF16 ? (void*)((unsigned short*)Cv + (size_t)bz * sC)
                     : (void*)((float*)Cv + (size_t)bz * sC);
    const int col0 = n0 + wn + lr;
    const int col1 = n0 + wn + 16 + lr;
    const float bv0 = bias ? bias[col0] : 0.f;
    const float bv1 = bias ? bias[col1] : 0.f;
    const int r0 = m0 + wm + half * 8;
    const int r1 = m0 + wm + 16 + half * 8;
    store_tile<OBF16>(Cb, ldc, r0, col0, acc00, scl, bv0);
    store_tile<OBF16>(Cb, ldc, r0, col1, acc01, scl, bv1);
    store_tile<OBF16>(Cb, ldc, r1, col0, acc10, scl, bv0);
    store_tile<OBF16>(Cb, ldc, r1, col1, acc11, scl, bv1);
}

extern "C" void kernel_launch(void* const* d_in, const int* in_sizes, int n_in,
                              void* d_out, int out_size, void* d_ws, size_t ws_size,
                              hipStream_t stream) {
    (void)in_sizes; (void)n_in; (void)out_size;
    const int B = 16, N = 2048, RNN = 128, INTER = 256;

    const float* flows = (const float*)d_in[0];
    const float* W1    = (const float*)d_in[1];
    const float* b1    = (const float*)d_in[2];
    const float* W2    = (const float*)d_in[3];
    const float* b2    = (const float*)d_in[4];
    const float* W3    = (const float*)d_in[5];
    const float* b3    = (const float*)d_in[6];
    float* out = (float*)d_out;

    // ---- workspace carve-out (256B-aligned slots) ----
    char* base = (char*)d_ws;
    size_t off = 0;
    auto take = [&](size_t bytes) -> char* {
        char* p = base + off;
        off = (off + bytes + 255) & ~(size_t)255;
        return p;
    };
    float*          dinv = (float*)take((size_t)B * N * 4);
    unsigned short* W1b  = (unsigned short*)take((size_t)N * RNN * 2);
    unsigned short* W2b  = (unsigned short*)take((size_t)RNN * INTER * 2);
    unsigned short* W3b  = (unsigned short*)take((size_t)INTER * RNN * 2);
    unsigned short* msg  = (unsigned short*)take((size_t)B * N * INTER * 2);
    unsigned short* xb   = (unsigned short*)take((size_t)B * N * INTER * 2);
    unsigned short* At   = (unsigned short*)take((size_t)B * N * N * 2);
    // main path (bf16 A^T copy of flows) needs ~170 MB; deterministic fallback
    // to transposed-f32 staging if the harness gave us less.
    const bool useAt = (ws_size >= off);

    const size_t NN = (size_t)N * N;

    degree_rsqrt_kernel<<<dim3(N / 256, B), 256, 0, stream>>>(flows, dinv, N);
    f32_to_bf16_kernel<<<(N * RNN + 255) / 256, 256, 0, stream>>>(W1, W1b, N * RNN);
    f32_to_bf16_kernel<<<(RNN * INTER + 255) / 256, 256, 0, stream>>>(W2, W2b, RNN * INTER);
    f32_to_bf16_kernel<<<(INTER * RNN + 255) / 256, 256, 0, stream>>>(W3, W3b, INTER * RNN);
    if (useAt)
        transpose_to_bf16_kernel<<<dim3(N / 32, N / 32, B), 256, 0, stream>>>(flows, At, N);

    // ---- Layer 1 (K = 2048, F = 128) ----
    gemm_rowscale_bias<0, true><<<dim3(RNN / 128, N / 64, B), 256, 0, stream>>>(
        flows, NN, N, W1b, 0, RNN, dinv, N, nullptr,
        msg, (size_t)N * RNN, RNN, N);                       // msg = dinv .* (X W1)
    if (useAt)
        gemm_rowscale_bias<2, true><<<dim3(RNN / 128, N / 64, B), 256, 0, stream>>>(
            At, NN, N, msg, (size_t)N * RNN, RNN, dinv, N, b1,
            xb, (size_t)N * RNN, RNN, N);                    // x1 = dinv .* (A^T msg) + b1
    else
        gemm_rowscale_bias<1, true><<<dim3(RNN / 128, N / 64, B), 256, 0, stream>>>(
            flows, NN, N, msg, (size_t)N * RNN, RNN, dinv, N, b1,
            xb, (size_t)N * RNN, RNN, N);

    // ---- Layer 2 (K = 128 then 2048, F = 256) ----
    gemm_rowscale_bias<2, true><<<dim3(INTER / 128, N / 64, B), 256, 0, stream>>>(
        xb, (size_t)N * RNN, RNN, W2b, 0, INTER, dinv, N, nullptr,
        msg, (size_t)N * INTER, INTER, RNN);
    if (useAt)
        gemm_rowscale_bias<2, true><<<dim3(INTER / 128, N / 64, B), 256, 0, stream>>>(
            At, NN, N, msg, (size_t)N * INTER, INTER, dinv, N, b2,
            xb, (size_t)N * INTER, INTER, N);                // x2 (overwrites x1; serialized)
    else
        gemm_rowscale_bias<1, true><<<dim3(INTER / 128, N / 64, B), 256, 0, stream>>>(
            flows, NN, N, msg, (size_t)N * INTER, INTER, dinv, N, b2,
            xb, (size_t)N * INTER, INTER, N);

    // ---- Layer 3 (K = 256 then 2048, F = 128) -> d_out (fp32) ----
    gemm_rowscale_bias<2, true><<<dim3(RNN / 128, N / 64, B), 256, 0, stream>>>(
        xb, (size_t)N * INTER, INTER, W3b, 0, RNN, dinv, N, nullptr,
        msg, (size_t)N * RNN, RNN, INTER);
    if (useAt)
        gemm_rowscale_bias<2, false><<<dim3(RNN / 128, N / 64, B), 256, 0, stream>>>(
            At, NN, N, msg, (size_t)N * RNN, RNN, dinv, N, b3,
            out, (size_t)N * RNN, RNN, N);
    else
        gemm_rowscale_bias<1, false><<<dim3(RNN / 128, N / 64, B), 256, 0, stream>>>(
            flows, NN, N, msg, (size_t)N * RNN, RNN, dinv, N, b3,
            out, (size_t)N * RNN, RNN, N);
}
